// MambaBlock_56212531970052
// MI455X (gfx1250) — matrix operand order repaired and checked by reading, not compiled, not run on hardware
//
#include <hip/hip_runtime.h>
#include <hip/hip_bf16.h>
#include <math.h>
#include <stdint.h>

// ---- Problem constants (from the reference) ----
#define D_MODEL 1024
#define D_STATE 16
#define D_CONV  4
#define D_INNER 2048
#define DT_RANK 64
#define BATCH   2
#define SEQ     2048
#define MROWS   (BATCH * SEQ)            // 4096 token rows
#define XPROJ_N (DT_RANK + 2 * D_STATE)  // 96
#define BK      32                       // K-step per WMMA (bf16 K=32)
#define BM      64                       // macro-tile rows
#define BN      128                      // macro-tile cols

typedef __attribute__((ext_vector_type(16))) __bf16 v16bf;
typedef __attribute__((ext_vector_type(8)))  float  v8f;

__device__ __forceinline__ float silu_f(float v) {
    return v / (1.0f + expf(-v));
}

// LDS byte offset of a __shared__ pointer: addrspacecast(3->flat) puts the
// wave-relative LDS byte address in the low 32 bits (shared aperture lives in
// the high dword), which is exactly what async-to-LDS VDST wants.
__device__ __forceinline__ uint32_t lds_off(const void* p) {
    return (uint32_t)(uintptr_t)p;
}

// -----------------------------------------------------------------------------
// fp32-in / fp32-out GEMM, bf16 WMMA core, async-DMA double-buffered LDS:
//   C[M,N] = act( A[M,K] @ W[K,N] + bias[N] )
// Block: 128 threads = 4 waves, macro-tile 64x128, k-step 32.
// Waves in a 2x2 grid; each wave owns 32x64 = 2 A-frags x 4 B-frags
// -> 8 WMMAs per k-step (A-frags reused 4x, B-frags 2x).
// Tiles staged global->LDS with GLOBAL_LOAD_ASYNC_TO_LDS_B128 (ASYNCcnt);
// double-buffer selection is a uniform scalar add (no indexed-VGPR access).
// ACT: 0 = identity, 1 = softplus.
// -----------------------------------------------------------------------------
template <int ACT>
__global__ __launch_bounds__(128)
void wmma_gemm(const float* __restrict__ A, int lda,
               const float* __restrict__ W, int ldw,
               const float* __restrict__ bias,
               float* __restrict__ C, int ldc,
               int M, int N, int K)
{
    __shared__ __align__(16) float sA[2][BM * BK];   // [buf][row*BK + k]   16KB
    __shared__ __align__(16) float sB[2][BK * BN];   // [buf][k*BN + col]   32KB

    const int tid   = threadIdx.x;
    const int lane  = tid & 31;
    const int wv    = tid >> 5;          // 0..3
    const int hf    = lane >> 4;         // half-wave select
    const int l16   = lane & 15;
    const int waveM = (wv >> 1) * 32;    // 0 / 32
    const int waveN = (wv & 1) * 64;     // 0 / 64

    const int tileM = blockIdx.y * BM;
    const int tileN = blockIdx.x * BN;

    // ---- Staging addresses. Each lane moves 16B per issue:
    //   A tile 64x32 = 2048 floats -> 4 issues; B tile 32x128 = 4096 -> 8.
    uint64_t gA0[4], gB0[8];
    uint32_t lA0[4], lB0[8];
#pragma unroll
    for (int it = 0; it < 4; ++it) {
        const int f  = it * 512 + tid * 4;
        const int aR = f >> 5, aC = f & (BK - 1);
        gA0[it] = (uint64_t)(uintptr_t)(A + (size_t)(tileM + aR) * lda + aC);
        lA0[it] = lds_off(&sA[0][f]);
    }
#pragma unroll
    for (int it = 0; it < 8; ++it) {
        const int f  = it * 512 + tid * 4;
        const int bK = f >> 7, bC = f & (BN - 1);
        int bCg = tileN + bC;
        if (bCg >= N) bCg = 0;   // clamp: junk cols never stored (epilogue guard)
        gB0[it] = (uint64_t)(uintptr_t)(W + (size_t)bK * ldw + bCg);
        lB0[it] = lds_off(&sB[0][f]);
    }

    // Async DMA of one (A,B) k-slice into LDS buffer `buf`.
    auto stage = [&](int buf, int s) {
        const uint64_t aoff = (uint64_t)s * (BK * 4);                 // bytes
        const uint64_t boff = (uint64_t)s * ((uint64_t)BK * ldw * 4); // bytes
        const uint32_t abuf = (uint32_t)buf * (BM * BK * 4);          // buffer hop
        const uint32_t bbuf = (uint32_t)buf * (BK * BN * 4);
#pragma unroll
        for (int it = 0; it < 4; ++it)
            asm volatile("global_load_async_to_lds_b128 %0, %1, off"
                         :: "v"(lA0[it] + abuf), "v"(gA0[it] + aoff) : "memory");
#pragma unroll
        for (int it = 0; it < 8; ++it)
            asm volatile("global_load_async_to_lds_b128 %0, %1, off"
                         :: "v"(lB0[it] + bbuf), "v"(gB0[it] + boff) : "memory");
    };

    v8f acc[2][4] = {};
    const int nsteps = K / BK;

    stage(0, 0);
    asm volatile("s_wait_asynccnt 0x0" ::: "memory");
    __syncthreads();

    for (int s = 0; s < nsteps; ++s) {
        const int buf = s & 1;
        if (s + 1 < nsteps) stage(buf ^ 1, s + 1);   // overlap DMA with math

        // A fragments: 16x32 bf16 (MxK), contiguous k = 16*hf + s per lane (B uses the same).
        v16bf afrag[2];
#pragma unroll
        for (int mt = 0; mt < 2; ++mt) {
            const float* ar = &sA[buf][(waveM + mt * 16 + l16) * BK];
#pragma unroll
            for (int v = 0; v < 8; ++v) {
                const int kb = hf * 16 + 2 * v;
                afrag[mt][2 * v]     = (__bf16)ar[kb];
                afrag[mt][2 * v + 1] = (__bf16)ar[kb + 1];
            }
        }
        // B fragments: 32x16 bf16 (KxN).
        v16bf bfrag[4];
#pragma unroll
        for (int nt = 0; nt < 4; ++nt) {
            const int coll = waveN + nt * 16 + l16;
#pragma unroll
            for (int v = 0; v < 8; ++v) {
                const int kb = hf * 16 + 2 * v;
                bfrag[nt][2 * v]     = (__bf16)sB[buf][kb * BN + coll];
                bfrag[nt][2 * v + 1] = (__bf16)sB[buf][(kb + 1) * BN + coll];
            }
        }
#pragma unroll
        for (int mt = 0; mt < 2; ++mt)
#pragma unroll
            for (int nt = 0; nt < 4; ++nt)
                acc[mt][nt] = __builtin_amdgcn_wmma_f32_16x16x32_bf16(
                    false, afrag[mt], false, bfrag[nt],
                    (short)0, acc[mt][nt], false, false);

        asm volatile("s_wait_asynccnt 0x0" ::: "memory");
        __syncthreads();
    }

    // Epilogue. D layout: lanes 0-15 -> M=r, lanes 16-31 -> M=r+8; N=l16.
#pragma unroll
    for (int mt = 0; mt < 2; ++mt)
#pragma unroll
        for (int nt = 0; nt < 4; ++nt) {
            const int colg = tileN + waveN + nt * 16 + l16;
            if (colg >= N) continue;            // divergence only at stores
            const float bval = bias ? bias[colg] : 0.0f;
#pragma unroll
            for (int r = 0; r < 8; ++r) {
                const int mrow = tileM + waveM + mt * 16 + r + hf * 8;
                float v = acc[mt][nt][r] + bval;
                if (ACT == 1) v = (v > 20.0f) ? v : log1pf(expf(v));
                C[(long)mrow * ldc + colg] = v;
            }
        }
}

// -----------------------------------------------------------------------------
// Depthwise causal conv1d (K=4) over seq + bias + SiLU. One thread per (b,t,d).
// xr holds [x_val | res] with row stride 2*D_INNER; we read the x_val half.
// -----------------------------------------------------------------------------
__global__ __launch_bounds__(256)
void conv_silu_kernel(const float* __restrict__ xr,
                      const float* __restrict__ cw,
                      const float* __restrict__ cb,
                      float* __restrict__ xc)
{
    const long idx = (long)blockIdx.x * blockDim.x + threadIdx.x;
    const int  d   = (int)(idx & (D_INNER - 1));
    const long m   = idx >> 11;
    const int  t   = (int)(m & (SEQ - 1));
    float acc = cb[d];
#pragma unroll
    for (int j = 0; j < D_CONV; ++j) {
        const int back = D_CONV - 1 - j;
        if (t - back >= 0)
            acc += xr[(m - back) * (2L * D_INNER) + d] * cw[d * D_CONV + j];
    }
    xc[idx] = silu_f(acc);
}

// -----------------------------------------------------------------------------
// Selective scan: sequential over SEQ. One lane per (channel, state-n);
// 2 channels per wave32, 16 state lanes each; y_t via 4-level shfl_xor.
// Prefetch next timestep's operands to hide latency behind the serial chain.
// -----------------------------------------------------------------------------
__global__ __launch_bounds__(256)
void scan_kernel(const float* __restrict__ dtf,   // (MROWS, D_INNER)
                 const float* __restrict__ xdbl,  // (MROWS, 96): B@64, C@80
                 const float* __restrict__ xc,    // (MROWS, D_INNER)
                 const float* __restrict__ log_A, // (D_INNER, D_STATE)
                 float* __restrict__ ys)          // (MROWS, D_INNER)
{
    const int lane = threadIdx.x & 31;
    const int n    = lane & 15;
    const int sub  = lane >> 4;
    const int wv   = threadIdx.x >> 5;
    const int chan = (blockIdx.x * 8 + wv) * 2 + sub;   // 0..BATCH*D_INNER-1
    const int b    = chan >> 11;
    const int d    = chan & (D_INNER - 1);

    const float An = -expf(log_A[d * D_STATE + n]);
    float h = 0.0f;
    const long base = (long)b * SEQ;
    for (int t = 0; t < SEQ; ++t) {
        const long m = base + t;
        if (t + 1 < SEQ) {
            __builtin_prefetch(&dtf[(m + 1) * D_INNER + d], 0, 0);
            __builtin_prefetch(&xc [(m + 1) * D_INNER + d], 0, 0);
        }
        const float dt = dtf[m * D_INNER + d];
        const float xv = xc [m * D_INNER + d];
        const float Bn = xdbl[m * XPROJ_N + DT_RANK + n];
        const float Cn = xdbl[m * XPROJ_N + DT_RANK + D_STATE + n];
        const float dA = expf(dt * An);
        h = dA * h + (dt * xv) * Bn;
        float p = h * Cn;
        p += __shfl_xor(p, 1);
        p += __shfl_xor(p, 2);
        p += __shfl_xor(p, 4);
        p += __shfl_xor(p, 8);
        if (n == 0) ys[m * D_INNER + d] = p;
    }
}

// -----------------------------------------------------------------------------
// Gate: g = (ys + xc*D) * silu(res), written in place over ys.
// -----------------------------------------------------------------------------
__global__ __launch_bounds__(256)
void gate_kernel(float* __restrict__ ys,
                 const float* __restrict__ xc,
                 const float* __restrict__ xr,
                 const float* __restrict__ Dv)
{
    const long idx = (long)blockIdx.x * blockDim.x + threadIdx.x;
    const int  d   = (int)(idx & (D_INNER - 1));
    const long m   = idx >> 11;
    const float res = xr[m * (2L * D_INNER) + D_INNER + d];
    const float y   = ys[idx] + xc[idx] * Dv[d];
    ys[idx] = y * silu_f(res);
}

// -----------------------------------------------------------------------------
extern "C" void kernel_launch(void* const* d_in, const int* in_sizes, int n_in,
                              void* d_out, int out_size, void* d_ws, size_t ws_size,
                              hipStream_t stream)
{
    (void)in_sizes; (void)n_in; (void)out_size; (void)ws_size;
    const float* x        = (const float*)d_in[0];
    const float* in_w     = (const float*)d_in[1];
    const float* in_b     = (const float*)d_in[2];
    const float* conv_w   = (const float*)d_in[3];
    const float* conv_b   = (const float*)d_in[4];
    const float* xproj_w  = (const float*)d_in[5];
    const float* dtproj_w = (const float*)d_in[6];
    const float* dtproj_b = (const float*)d_in[7];
    const float* log_A    = (const float*)d_in[8];
    const float* Dvec     = (const float*)d_in[9];
    const float* out_w    = (const float*)d_in[10];
    const float* out_b    = (const float*)d_in[11];
    float* out = (float*)d_out;

    // Workspace carve-up (fp32 elements), ~169 MB total.
    float* ws   = (float*)d_ws;
    float* xr   = ws;                                   // MROWS * 2*D_INNER
    float* xc   = xr   + (long)MROWS * 2 * D_INNER;     // MROWS * D_INNER
    float* xdbl = xc   + (long)MROWS * D_INNER;         // MROWS * 96
    float* dtf  = xdbl + (long)MROWS * XPROJ_N;         // MROWS * D_INNER
    float* ys   = dtf  + (long)MROWS * D_INNER;         // MROWS * D_INNER

    const dim3 blk(128);

    // 1) in_proj: xr = x @ in_proj_w + b   (4096x1024x4096)
    wmma_gemm<0><<<dim3((2 * D_INNER) / BN, MROWS / BM), blk, 0, stream>>>(
        x, D_MODEL, in_w, 2 * D_INNER, in_b, xr, 2 * D_INNER,
        MROWS, 2 * D_INNER, D_MODEL);

    // 2) depthwise conv + SiLU -> xc
    conv_silu_kernel<<<(MROWS * D_INNER) / 256, 256, 0, stream>>>(xr, conv_w, conv_b, xc);

    // 3) x_dbl = xc @ x_proj_w   (4096x2048x96, no bias)
    wmma_gemm<0><<<dim3((XPROJ_N + BN - 1) / BN, MROWS / BM), blk, 0, stream>>>(
        xc, D_INNER, xproj_w, XPROJ_N, nullptr, xdbl, XPROJ_N,
        MROWS, XPROJ_N, D_INNER);

    // 4) dt = softplus(x_dbl[:, :64] @ dt_proj_w + b)   (4096x64x2048)
    wmma_gemm<1><<<dim3(D_INNER / BN, MROWS / BM), blk, 0, stream>>>(
        xdbl, XPROJ_N, dtproj_w, D_INNER, dtproj_b, dtf, D_INNER,
        MROWS, D_INNER, DT_RANK);

    // 5) selective scan -> ys
    scan_kernel<<<(BATCH * D_INNER / 2) / 8, 256, 0, stream>>>(dtf, xdbl, xc, log_A, ys);

    // 6) gate: ys = (ys + xc*D) * silu(res)   (in place)
    gate_kernel<<<(MROWS * D_INNER) / 256, 256, 0, stream>>>(ys, xc, xr, Dvec);

    // 7) out_proj: out = ys @ out_proj_w + b   (4096x2048x1024)
    wmma_gemm<0><<<dim3(D_MODEL / BN, MROWS / BM), blk, 0, stream>>>(
        ys, D_INNER, out_w, D_MODEL, out_b, out, D_MODEL,
        MROWS, D_MODEL, D_INNER);
}
